// InfoNCELoss_87290915323998
// MI455X (gfx1250) — compile-verified, matrix operand order checked by reading, not run on hardware
//
#include <hip/hip_runtime.h>
#include <math.h>

// ---------------------------------------------------------------------------
// InfoNCE loss, MI455X (gfx1250).
// GEMM path: split-bf16 (hi/lo) emulated-fp32 via V_WMMA_F32_16X16X32_BF16,
// fused with normalize/exp/row-sum (512MB sim matrix never materialized).
// 16-bit XDL WMMAs co-execute with the VALU exp epilogue (ISA §8.4).
// Error of the 3-product split ~2^-16 relative (fp32-comparable).
// ---------------------------------------------------------------------------

#define EPS 1e-6f

constexpr int BS       = 8192;     // batch
constexpr int D        = 128;      // feature dim
constexpr int COLS     = 2 * BS;   // cat = [noise; f]
constexpr int MT       = 16;       // WMMA M tile
constexpr int NT       = 16;       // WMMA N tile
constexpr int ROWTILES = BS / MT;     // 512
constexpr int COLTILES = COLS / NT;   // 1024
constexpr int COLSPLIT = 16;          // blocks along column dimension
constexpr int WPB      = 8;           // waves per 256-thread block (wave32)
constexpr int TPW      = COLTILES / (COLSPLIT * WPB);   // 8 col tiles / wave

typedef float  v4f  __attribute__((ext_vector_type(4)));
typedef float  v8f  __attribute__((ext_vector_type(8)));
typedef __bf16 v8bf __attribute__((ext_vector_type(8)));
typedef __bf16 v16bf __attribute__((ext_vector_type(16)));

union V16U { v16bf v; v8bf h[2]; };

static __device__ __forceinline__ unsigned short bf16_rne(float x) {
    unsigned int u = __float_as_uint(x);
    unsigned int r = u + 0x7FFFu + ((u >> 16) & 1u);   // round-to-nearest-even
    return (unsigned short)(r >> 16);
}
static __device__ __forceinline__ float bf16_as_f32(unsigned short h) {
    return __uint_as_float(((unsigned int)h) << 16);
}

// Load the 16 bf16 K-values one lane owns inside a 32-wide K window:
// elements [0..7] at p, elements [8..15] at p+16 (16B-aligned v8bf chunks).
static __device__ __forceinline__ v16bf load_bf16_pair(const unsigned short* p) {
    V16U t;
    t.h[0] = *(const v8bf*)(p);
    t.h[1] = *(const v8bf*)(p + 16);
    return t.v;
}

// -------------------------------------------------------------------- zero --
__global__ void infonce_zero_kernel(float* __restrict__ rowsum) {
    int i = blockIdx.x * blockDim.x + threadIdx.x;
    if (i < BS) rowsum[i] = 0.0f;
}

// ---------------------------------------------------- split fp32 -> hi/lo ---
// x ~= hi + lo exactly to ~2^-16 rel; hi = RNE_bf16(x), lo = RNE_bf16(x-hi).
__global__ void infonce_split_kernel(const float* __restrict__ f,
                                     const float* __restrict__ noise,
                                     unsigned short* __restrict__ fh,
                                     unsigned short* __restrict__ fl,
                                     unsigned short* __restrict__ nh,
                                     unsigned short* __restrict__ nl) {
    int i = blockIdx.x * blockDim.x + threadIdx.x;
    if (i >= BS * D) return;
    float x = f[i];
    unsigned short xh = bf16_rne(x);
    fh[i] = xh;
    fl[i] = bf16_rne(x - bf16_as_f32(xh));
    float y = noise[i];
    unsigned short yh = bf16_rne(y);
    nh[i] = yh;
    nl[i] = bf16_rne(y - bf16_as_f32(yh));
}

// ------------------------------------------------------- column norms nc[] --
// nc[j] = ||cat_j|| in exact fp32 (these feed every denominator).
__global__ void infonce_norms_kernel(const float* __restrict__ f,
                                     const float* __restrict__ noise,
                                     float* __restrict__ nc) {
    int gid  = blockIdx.x * blockDim.x + threadIdx.x;
    int wave = gid >> 5;
    int lane = gid & 31;
    if (wave >= COLS) return;
    const float* row = (wave < BS) ? (noise + (size_t)wave * D)
                                   : (f + (size_t)(wave - BS) * D);
    v4f v = *(const v4f*)(row + lane * 4);
    float s = v.x * v.x + v.y * v.y + v.z * v.z + v.w * v.w;
    s += __shfl_xor(s, 16, 32);
    s += __shfl_xor(s, 8, 32);
    s += __shfl_xor(s, 4, 32);
    s += __shfl_xor(s, 2, 32);
    s += __shfl_xor(s, 1, 32);
    if (lane == 0) nc[wave] = sqrtf(s);
}

// ------------------------------------------------- diagonal terms pos/self --
// Exact fp32: pos enters the numerator of the loss directly.
__global__ void infonce_diag_kernel(const float* __restrict__ f,
                                    const float* __restrict__ noise,
                                    float* __restrict__ pos_e,
                                    float* __restrict__ self_e) {
    int gid  = blockIdx.x * blockDim.x + threadIdx.x;
    int wave = gid >> 5;
    int lane = gid & 31;
    if (wave >= BS) return;
    v4f fv = *(const v4f*)(f + (size_t)wave * D + lane * 4);
    v4f nv = *(const v4f*)(noise + (size_t)wave * D + lane * 4);
    float dot = fv.x * nv.x + fv.y * nv.y + fv.z * nv.z + fv.w * nv.w;
    float ff  = fv.x * fv.x + fv.y * fv.y + fv.z * fv.z + fv.w * fv.w;
    float nn  = nv.x * nv.x + nv.y * nv.y + nv.z * nv.z + nv.w * nv.w;
#pragma unroll
    for (int m = 16; m >= 1; m >>= 1) {
        dot += __shfl_xor(dot, m, 32);
        ff  += __shfl_xor(ff, m, 32);
        nn  += __shfl_xor(nn, m, 32);
    }
    if (lane == 0) {
        float nfv = sqrtf(ff);
        float nnv = sqrtf(nn);
        pos_e[wave]  = __expf(dot / fmaxf(nfv * nnv, EPS));
        self_e[wave] = __expf(ff / fmaxf(nfv * nfv, EPS));
    }
}

// ------------------------- fused split-bf16 GEMM + exp + row-sum (WMMA) -----
// grid = (COLSPLIT, ROWTILES), block = 256 (8 wave32 waves).
// Per wave: one 16-row tile; A hi/lo register-resident (8 x v16bf = 64 VGPRs);
// per 16x16 col tile: 4 K-steps x 3 bf16 WMMAs (hi*hi + hi*lo + lo*hi),
// then sims = acc * rcp(max(nf*nc,eps)), exp, per-lane row partials;
// one 16-lane butterfly + global_atomic_add_f32 per row at the end.
__global__ __launch_bounds__(256) void infonce_gemm_kernel(
        const unsigned short* __restrict__ fh, const unsigned short* __restrict__ fl,
        const unsigned short* __restrict__ nh, const unsigned short* __restrict__ nl,
        const float* __restrict__ nc, float* __restrict__ rowsum) {
    const int lane = threadIdx.x & 31;
    const int wave = threadIdx.x >> 5;
    const int half = lane >> 4;     // 0: lanes 0-15, 1: lanes 16-31
    const int n    = lane & 15;     // N index (and A-row index m)
    const int row0 = blockIdx.y * MT;

    // A tile (16x128) hi/lo in registers. Lane's K chunks per 32-K window s:
    //   [s*32 + half*8 .. +7] and [s*32 + 16 + half*8 .. +7]
    const unsigned short* arh = fh + (size_t)(row0 + n) * D + half * 8;
    const unsigned short* arl = fl + (size_t)(row0 + n) * D + half * 8;
    v16bf aH[4], aL[4];
#pragma unroll
    for (int s = 0; s < 4; ++s) {
        aH[s] = load_bf16_pair(arh + s * 32);
        aL[s] = load_bf16_pair(arl + s * 32);
    }

    float nfv[8];
#pragma unroll
    for (int v = 0; v < 8; ++v) nfv[v] = nc[BS + row0 + v + 8 * half];

    float rsum[8] = {0.f, 0.f, 0.f, 0.f, 0.f, 0.f, 0.f, 0.f};

    for (int t = 0; t < TPW; ++t) {
        const int jt = blockIdx.x * WPB + wave + t * (COLSPLIT * WPB);
        const bool   in_noise = (jt < COLTILES / 2);
        const size_t boff = in_noise ? (size_t)jt * NT * D
                                     : (size_t)(jt - COLTILES / 2) * NT * D;
        const unsigned short* bh = (in_noise ? nh : fh) + boff + (size_t)n * D + half * 8;
        const unsigned short* bl = (in_noise ? nl : fl) + boff + (size_t)n * D + half * 8;

        // Prefetch next tile's B stream while this tile computes.
        if (t + 1 < TPW) {
            const int jn = jt + COLSPLIT * WPB;
            const bool pn = (jn < COLTILES / 2);
            const size_t poff = pn ? (size_t)jn * NT * D
                                   : (size_t)(jn - COLTILES / 2) * NT * D;
            __builtin_prefetch((pn ? nh : fh) + poff + (size_t)n * D, 0, 1);
            __builtin_prefetch((pn ? nl : fl) + poff + (size_t)n * D, 0, 1);
        }

        v8f acc = {};
#pragma unroll
        for (int s = 0; s < 4; ++s) {
            v16bf bH = load_bf16_pair(bh + s * 32);
            v16bf bL = load_bf16_pair(bl + s * 32);
            acc = __builtin_amdgcn_wmma_f32_16x16x32_bf16(
                false, aH[s], false, bH, (short)0, acc, false, false);
            acc = __builtin_amdgcn_wmma_f32_16x16x32_bf16(
                false, aH[s], false, bL, (short)0, acc, false, false);
            acc = __builtin_amdgcn_wmma_f32_16x16x32_bf16(
                false, aL[s], false, bH, (short)0, acc, false, false);
        }

        const float ncv = nc[jt * NT + n];
#pragma unroll
        for (int v = 0; v < 8; ++v) {
            float invd = __builtin_amdgcn_rcpf(fmaxf(nfv[v] * ncv, EPS));
            rsum[v] += __expf(acc[v] * invd);
        }
    }

#pragma unroll
    for (int v = 0; v < 8; ++v) {
        float s = rsum[v];
        s += __shfl_xor(s, 1, 32);
        s += __shfl_xor(s, 2, 32);
        s += __shfl_xor(s, 4, 32);
        s += __shfl_xor(s, 8, 32);
        if (n == 0) atomicAdd(&rowsum[row0 + v + 8 * half], s);
    }
}

// ----------------------------------------------------------- final reduce ---
__global__ void infonce_final_kernel(const float* __restrict__ rowsum,
                                     const float* __restrict__ pos_e,
                                     const float* __restrict__ self_e,
                                     float* __restrict__ out) {
    __shared__ float sdata[256];
    float s = 0.0f;
    for (int i = threadIdx.x; i < BS; i += 256) {
        float pos  = pos_e[i];
        float self = self_e[i];
        float neg  = rowsum[i] - pos - self;
        s += -logf(pos / (neg + EPS));
    }
    sdata[threadIdx.x] = s;
    __syncthreads();
#pragma unroll
    for (int st = 128; st > 0; st >>= 1) {
        if (threadIdx.x < st) sdata[threadIdx.x] += sdata[threadIdx.x + st];
        __syncthreads();
    }
    if (threadIdx.x == 0) out[0] = sdata[0] / (float)BS;
}

// ---------------------------------------------------------------------------
extern "C" void kernel_launch(void* const* d_in, const int* in_sizes, int n_in,
                              void* d_out, int out_size, void* d_ws, size_t ws_size,
                              hipStream_t stream) {
    const float* f     = (const float*)d_in[0];
    const float* noise = (const float*)d_in[1];
    float* out = (float*)d_out;

    // Scratch layout:
    //   floats : nc[16384] | rowsum[8192] | pos_e[8192] | self_e[8192]
    //   ushort : fh[1M] | fl[1M] | nh[1M] | nl[1M]   (bf16 hi/lo planes, 8MB)
    float* nc     = (float*)d_ws;
    float* rowsum = nc + COLS;
    float* pos_e  = rowsum + BS;
    float* self_e = pos_e + BS;
    unsigned short* fh = (unsigned short*)(self_e + BS);
    unsigned short* fl = fh + (size_t)BS * D;
    unsigned short* nh = fl + (size_t)BS * D;
    unsigned short* nl = nh + (size_t)BS * D;

    infonce_zero_kernel<<<BS / 256, 256, 0, stream>>>(rowsum);
    infonce_split_kernel<<<(BS * D) / 256, 256, 0, stream>>>(f, noise, fh, fl, nh, nl);
    infonce_norms_kernel<<<(COLS * 32) / 256, 256, 0, stream>>>(f, noise, nc);
    infonce_diag_kernel<<<(BS * 32) / 256, 256, 0, stream>>>(f, noise, pos_e, self_e);
    infonce_gemm_kernel<<<dim3(COLSPLIT, ROWTILES), 256, 0, stream>>>(fh, fl, nh, nl, nc, rowsum);
    infonce_final_kernel<<<1, 256, 0, stream>>>(rowsum, pos_e, self_e, out);
}
